// BiBAECPI_36051955483142
// MI455X (gfx1250) — matrix operand order repaired and checked by reading, not compiled, not run on hardware
//
#include <hip/hip_runtime.h>
#include <hip/hip_bf16.h>

// ---------------------------------------------------------------------------
// BiBAECPI forward for MI455X (gfx1250, wave32, WMMA).
// Big GEMMs (GIN MLPs, protein fc, conv1d-as-GEMM) -> v_wmma_f32_16x16x32_bf16.
// Staging via GLOBAL_LOAD_ASYNC_TO_LDS_B128 (ASYNCcnt) when available.
// Everything else (norms, scatter, GLU, head) -> f32 VALU.
// ---------------------------------------------------------------------------

#define NN      8192      // N_NODES
#define NE      131072    // N_EDGES
#define LP      8192      // L_PROT
#define COMP    256
#define PROT    256
#define HID     512
#define LAT     512
#define KHEAD   128
#define KSZ     11
#define CPAD    5
#define EPSN    1e-5f
#define SCALE_R 0.70710678118654752f

typedef __attribute__((ext_vector_type(16))) __bf16 v16bf;
typedef __attribute__((ext_vector_type(8)))  float  v8f;

struct Frag32 { uint4 q[2]; };   // 32 bytes == v16bf

// Optional CDNA5 async global->LDS staging (ASYNCcnt path).
// Probe-confirmed signature: (int4_AS1*, int4_AS3*, imm offset, imm cpol)
// where int4 is a GCC-style vector_size(16) int vector.
#if defined(__has_builtin)
#  if __has_builtin(__builtin_amdgcn_global_load_async_to_lds_b128)
#    define USE_ASYNC_LDS 1
#  endif
#endif
#ifndef USE_ASYNC_LDS
#  define USE_ASYNC_LDS 0
#endif

#if USE_ASYNC_LDS
typedef int v4i_gcc __attribute__((vector_size(16)));
typedef __attribute__((address_space(1))) v4i_gcc glob_v4i;
typedef __attribute__((address_space(3))) v4i_gcc lds_v4i;
#endif

__device__ __forceinline__ void wait_asynccnt0() {
#if defined(__has_builtin)
#  if __has_builtin(__builtin_amdgcn_s_wait_asynccnt)
  __builtin_amdgcn_s_wait_asynccnt(0);
  return;
#  endif
#endif
  asm volatile("s_wait_asynccnt 0x0" ::: "memory");
}

__device__ __forceinline__ unsigned short f2bf(float f) {
  unsigned int u = __builtin_bit_cast(unsigned int, f);
  u += 0x7FFFu + ((u >> 16) & 1u);   // round-to-nearest-even
  return (unsigned short)(u >> 16);
}

// ---------------------------------------------------------------------------
// WMMA GEMM:  C[M,N] = A[M,K](bf16,row-major,lda) * W[N,K](bf16,row-major,ldw)^T
//             (+ bias[N] when !accumulate, += C when accumulate)
// block: 256 thr = 8 waves (4 in M x 2 in N); block tile 128x64;
// wave tile 32x32 -> 4 accumulators, 4 WMMA per K-step.
// grid: (N/64, M/128). Requires M%128==0, N%64==0, K%32==0.
// ---------------------------------------------------------------------------
__global__ __launch_bounds__(256)
void wmma_gemm_bf16(const unsigned short* __restrict__ A, int lda,
                    const unsigned short* __restrict__ W, int ldw,
                    float* __restrict__ C, int ldc,
                    const float* __restrict__ bias,
                    int K, int accumulate)
{
  __shared__ unsigned short As[128][32];
  __shared__ unsigned short Ws[64][32];

  const int tid  = threadIdx.x;
  const int wave = tid >> 5;
  const int lane = tid & 31;
  const int half = lane >> 4;
  const int ln   = lane & 15;
  const int m0   = blockIdx.y * 128;
  const int n0   = blockIdx.x * 64;
  const int wm   = (wave >> 1) * 32;   // wave row offset inside block tile
  const int wn   = (wave & 1) * 32;    // wave col offset inside block tile

  v8f acc[2][2];
#pragma unroll
  for (int a = 0; a < 2; ++a)
#pragma unroll
    for (int b = 0; b < 2; ++b)
#pragma unroll
      for (int r = 0; r < 8; ++r) {
        int row = m0 + wm + a * 16 + half * 8 + r;
        int col = n0 + wn + b * 16 + ln;
        acc[a][b][r] = accumulate ? C[(size_t)row * ldc + col]
                                  : (bias ? bias[col] : 0.f);
      }

  for (int k0 = 0; k0 < K; k0 += 32) {
#if USE_ASYNC_LDS
    // stage A (128x32 = 512 xfers, 2/thread) + W (64x32 = 256 xfers, 1/thread)
    // via GLOBAL_LOAD_ASYNC_TO_LDS_B128 (tracked with ASYNCcnt)
#pragma unroll
    for (int i = 0; i < 2; ++i) {
      int s = tid + i * 256;
      int row = s >> 2, quad = s & 3;
      __builtin_amdgcn_global_load_async_to_lds_b128(
          (glob_v4i*)(A + (size_t)(m0 + row) * lda + k0 + quad * 8),
          (lds_v4i*)&As[row][quad * 8], 0, 0);
    }
    {
      int row = tid >> 2, quad = tid & 3;
      __builtin_amdgcn_global_load_async_to_lds_b128(
          (glob_v4i*)(W + (size_t)(n0 + row) * ldw + k0 + quad * 8),
          (lds_v4i*)&Ws[row][quad * 8], 0, 0);
    }
    wait_asynccnt0();
#else
    // stage A tile: 128x32 bf16 = 512 uint4 (2 per thread)
#pragma unroll
    for (int i = 0; i < 2; ++i) {
      int s = tid + i * 256;
      int row = s >> 2, quad = s & 3;
      *reinterpret_cast<uint4*>(&As[row][quad * 8]) =
        *reinterpret_cast<const uint4*>(A + (size_t)(m0 + row) * lda + k0 + quad * 8);
    }
    // stage W tile: 64x32 bf16 = 256 uint4 (1 per thread)
    {
      int row = tid >> 2, quad = tid & 3;
      *reinterpret_cast<uint4*>(&Ws[row][quad * 8]) =
        *reinterpret_cast<const uint4*>(W + (size_t)(n0 + row) * ldw + k0 + quad * 8);
    }
#endif
    if (k0 + 32 < K)  // gfx1250 global_prefetch_b8 on next K tile
      __builtin_prefetch(A + (size_t)(m0 + (tid >> 1)) * lda + k0 + 32, 0, 1);
    __syncthreads();

    // A fragments: lane=row, K runs [half*8,+8) and [16+half*8,+8)
    Frag32 fa0, fa1;
    fa0.q[0] = *reinterpret_cast<const uint4*>(&As[wm + ln][half * 8]);
    fa0.q[1] = *reinterpret_cast<const uint4*>(&As[wm + ln][16 + half * 8]);
    fa1.q[0] = *reinterpret_cast<const uint4*>(&As[wm + 16 + ln][half * 8]);
    fa1.q[1] = *reinterpret_cast<const uint4*>(&As[wm + 16 + ln][16 + half * 8]);
    v16bf a0 = __builtin_bit_cast(v16bf, fa0);
    v16bf a1 = __builtin_bit_cast(v16bf, fa1);
    // B fragments: lane=col, K runs [half*8,+8) and [16+half*8,+8)
    Frag32 fb0, fb1;
    fb0.q[0] = *reinterpret_cast<const uint4*>(&Ws[wn + ln][half * 8]);
    fb0.q[1] = *reinterpret_cast<const uint4*>(&Ws[wn + ln][16 + half * 8]);
    fb1.q[0] = *reinterpret_cast<const uint4*>(&Ws[wn + 16 + ln][half * 8]);
    fb1.q[1] = *reinterpret_cast<const uint4*>(&Ws[wn + 16 + ln][16 + half * 8]);
    v16bf b0 = __builtin_bit_cast(v16bf, fb0);
    v16bf b1 = __builtin_bit_cast(v16bf, fb1);

    acc[0][0] = __builtin_amdgcn_wmma_f32_16x16x32_bf16(false, a0, false, b0,
                                                        (short)0, acc[0][0], false, false);
    acc[0][1] = __builtin_amdgcn_wmma_f32_16x16x32_bf16(false, a0, false, b1,
                                                        (short)0, acc[0][1], false, false);
    acc[1][0] = __builtin_amdgcn_wmma_f32_16x16x32_bf16(false, a1, false, b0,
                                                        (short)0, acc[1][0], false, false);
    acc[1][1] = __builtin_amdgcn_wmma_f32_16x16x32_bf16(false, a1, false, b1,
                                                        (short)0, acc[1][1], false, false);
    __syncthreads();
  }

#pragma unroll
  for (int a = 0; a < 2; ++a)
#pragma unroll
    for (int b = 0; b < 2; ++b)
#pragma unroll
      for (int r = 0; r < 8; ++r) {
        int row = m0 + wm + a * 16 + half * 8 + r;
        int col = n0 + wn + b * 16 + ln;
        C[(size_t)row * ldc + col] = acc[a][b][r];
      }
}

// ---------------------------------------------------------------------------
// Elementwise / data-movement helpers
// ---------------------------------------------------------------------------
__global__ void copy_f32_k(const float* __restrict__ in, float* __restrict__ out, size_t n) {
  size_t i = (size_t)blockIdx.x * blockDim.x + threadIdx.x;
  size_t s = (size_t)gridDim.x * blockDim.x;
  for (; i < n; i += s) out[i] = in[i];
}
__global__ void f32_to_bf16_k(const float* __restrict__ in, unsigned short* __restrict__ out, size_t n) {
  size_t i = (size_t)blockIdx.x * blockDim.x + threadIdx.x;
  size_t s = (size_t)gridDim.x * blockDim.x;
  for (; i < n; i += s) out[i] = f2bf(in[i]);
}
__global__ void zero_u16_k(unsigned short* __restrict__ p, size_t n) {
  size_t i = (size_t)blockIdx.x * blockDim.x + threadIdx.x;
  size_t s = (size_t)gridDim.x * blockDim.x;
  for (; i < n; i += s) p[i] = 0;
}
__global__ void zero_f32_k(float* __restrict__ p, size_t n) {
  size_t i = (size_t)blockIdx.x * blockDim.x + threadIdx.x;
  size_t s = (size_t)gridDim.x * blockDim.x;
  for (; i < n; i += s) p[i] = 0.f;
}

__global__ void gather_rows_f32_k(const float* __restrict__ emb, const int* __restrict__ idx,
                                  float* __restrict__ out, int D) {
  int r = blockIdx.x;
  const float* src = emb + (size_t)idx[r] * D;
  for (int c = threadIdx.x; c < D; c += blockDim.x) out[(size_t)r * D + c] = src[c];
}
__global__ void gather_rows_bf16_k(const float* __restrict__ emb, const int* __restrict__ idx,
                                   unsigned short* __restrict__ out, int D) {
  int r = blockIdx.x;
  const float* src = emb + (size_t)idx[r] * D;
  for (int c = threadIdx.x; c < D; c += blockDim.x) out[(size_t)r * D + c] = f2bf(src[c]);
}

// z[dst] += h[src]  (segment_sum): one block per edge, global f32 atomics
__global__ void edge_scatter_k(const float* __restrict__ h, float* __restrict__ z,
                               const int* __restrict__ src, const int* __restrict__ dst, int D) {
  int e = blockIdx.x;
  size_t sb = (size_t)src[e] * D, db = (size_t)dst[e] * D;
  for (int c = threadIdx.x; c < D; c += blockDim.x)
    atomicAdd(&z[db + c], h[sb + c]);
}

// pack conv weights: w[co][ci][k] f32 -> out[k][co][ci] bf16
__global__ void pack_conv_k(const float* __restrict__ w, unsigned short* __restrict__ out,
                            int CO, int CI, int KS) {
  size_t total = (size_t)CO * CI * KS;
  size_t i = (size_t)blockIdx.x * blockDim.x + threadIdx.x;
  size_t s = (size_t)gridDim.x * blockDim.x;
  for (; i < total; i += s) {
    int k = (int)(i % KS);
    size_t t = i / KS;
    int ci = (int)(t % CI);
    int co = (int)(t / CI);
    out[((size_t)k * CO + co) * CI + ci] = f2bf(w[i]);
  }
}

// ---------------------------------------------------------------------------
// Row-wise instance-norm variants: one wave32 per row, grid.x = rows/8
// ---------------------------------------------------------------------------
__device__ __forceinline__ void wave_msum(float& s, float& s2) {
#pragma unroll
  for (int off = 16; off; off >>= 1) {
    s  += __shfl_xor(s,  off);
    s2 += __shfl_xor(s2, off);
  }
}

// relu(inorm(x)); optional second relu(inorm(.)) pass; f32 and/or bf16 outputs
__global__ __launch_bounds__(256)
void inorm_relu_k(const float* __restrict__ in, float* __restrict__ outF,
                  unsigned short* __restrict__ outB, int D, int twice) {
  int row  = blockIdx.x * 8 + (threadIdx.x >> 5);
  int lane = threadIdx.x & 31;
  int n = D >> 5;                       // 8 or 16 elements per lane
  const float* rp = in + (size_t)row * D;
  float v[16];
  float s = 0.f, s2 = 0.f;
#pragma unroll 16
  for (int i = 0; i < 16; ++i) if (i < n) {
    float x = rp[lane + i * 32];
    v[i] = x; s += x; s2 += x * x;
  }
  wave_msum(s, s2);
  float mean = s / D, var = s2 / D - mean * mean;
  float rs = rsqrtf(var + EPSN);
  s = 0.f; s2 = 0.f;
#pragma unroll 16
  for (int i = 0; i < 16; ++i) if (i < n) {
    float x = fmaxf((v[i] - mean) * rs, 0.f);
    v[i] = x; s += x; s2 += x * x;
  }
  if (twice) {
    wave_msum(s, s2);
    mean = s / D; var = s2 / D - mean * mean;
    rs = rsqrtf(var + EPSN);
#pragma unroll 16
    for (int i = 0; i < 16; ++i) if (i < n)
      v[i] = fmaxf((v[i] - mean) * rs, 0.f);
  }
#pragma unroll 16
  for (int i = 0; i < 16; ++i) if (i < n) {
    size_t o = (size_t)row * D + lane + i * 32;
    if (outF) outF[o] = v[i];
    if (outB) outB[o] = f2bf(v[i]);
  }
}

// inorm(leaky_relu(x,0.1)) then atomic column-sum into out[D]
__global__ __launch_bounds__(256)
void leaky_inorm_colsum_k(const float* __restrict__ in, float* __restrict__ out, int D) {
  int row  = blockIdx.x * 8 + (threadIdx.x >> 5);
  int lane = threadIdx.x & 31;
  int n = D >> 5;
  const float* rp = in + (size_t)row * D;
  float v[16];
  float s = 0.f, s2 = 0.f;
#pragma unroll 16
  for (int i = 0; i < 16; ++i) if (i < n) {
    float x = rp[lane + i * 32];
    x = x > 0.f ? x : 0.1f * x;
    v[i] = x; s += x; s2 += x * x;
  }
  wave_msum(s, s2);
  float mean = s / D, rs = rsqrtf(s2 / D - mean * mean + EPSN);
#pragma unroll 16
  for (int i = 0; i < 16; ++i) if (i < n)
    atomicAdd(&out[lane + i * 32], (v[i] - mean) * rs);
}

// column sum: grid.x = D, block reduces over rows
__global__ __launch_bounds__(256)
void colsum_k(const float* __restrict__ in, float* __restrict__ out, int rows, int D) {
  __shared__ float red[256];
  int c = blockIdx.x;
  float s = 0.f;
  for (int r = threadIdx.x; r < rows; r += blockDim.x) s += in[(size_t)r * D + c];
  red[threadIdx.x] = s;
  __syncthreads();
  for (int off = 128; off; off >>= 1) {
    if (threadIdx.x < off) red[threadIdx.x] += red[threadIdx.x + off];
    __syncthreads();
  }
  if (threadIdx.x == 0) out[c] = red[0];
}

// y[o] = (init?0:y[o]) + b[o] + sum_k x[k]*W[o*Kk+k]
__global__ void matvec_k(const float* __restrict__ x, const float* __restrict__ W,
                         const float* __restrict__ b, float* __restrict__ y,
                         int Kk, int O, int init) {
  int o = blockIdx.x * blockDim.x + threadIdx.x;
  if (o >= O) return;
  float s = init ? 0.f : y[o];
  if (b) s += b[o];
  for (int k = 0; k < Kk; ++k) s += x[k] * W[(size_t)o * Kk + k];
  y[o] = s;
}

// GLU + residual: x = (u*sigmoid(g) + x)*sqrt(0.5); also refresh padded bf16
__global__ void glu_residual_k(const float* __restrict__ y, float* __restrict__ x,
                               unsigned short* __restrict__ xpadb) {
  int l = blockIdx.x;
  for (int c = threadIdx.x; c < LAT; c += blockDim.x) {
    float u = y[(size_t)l * (2 * LAT) + c];
    float g = y[(size_t)l * (2 * LAT) + LAT + c];
    float v = (u * (1.f / (1.f + __expf(-g))) + x[(size_t)l * LAT + c]) * SCALE_R;
    x[(size_t)l * LAT + c] = v;
    xpadb[(size_t)(l + CPAD) * LAT + c] = f2bf(v);
  }
}

// ---------------------------------------------------------------------------
// Fusion head: 1 block, 512 threads
// ---------------------------------------------------------------------------
__global__ __launch_bounds__(512)
void head_k(const float* __restrict__ score, const float* __restrict__ aminoVec,
            const float* __restrict__ fps,
            const float* __restrict__ tc_w, const float* __restrict__ tc_b,
            const float* __restrict__ tp_w, const float* __restrict__ tp_b,
            const float* __restrict__ mu_theta, const float* __restrict__ mu_beta,
            const float* __restrict__ fp0, const float* __restrict__ fp1,
            const float* __restrict__ out_w, const float* __restrict__ out_b,
            const int* __restrict__ c_id, const int* __restrict__ p_id,
            float* __restrict__ out) {
  __shared__ float av[LAT];
  __shared__ float red[LAT];
  __shared__ float fv1[LAT];
  __shared__ float fusion[3 * KHEAD];
  int t = threadIdx.x;

  float x = score[t];
  red[t] = x; __syncthreads();
  for (int off = 256; off; off >>= 1) { if (t < off) red[t] += red[t + off]; __syncthreads(); }
  float mean = red[0] / (float)LAT; __syncthreads();
  red[t] = x * x; __syncthreads();
  for (int off = 256; off; off >>= 1) { if (t < off) red[t] += red[t + off]; __syncthreads(); }
  float var = red[0] / (float)LAT - mean * mean;
  av[t] = (x - mean) * rsqrtf(var + EPSN);
  __syncthreads();

  if (t < KHEAD) {
    float ck = tc_b[t], pk = tp_b[t];
    for (int k = 0; k < LAT; ++k) {
      ck += av[k]       * tc_w[(size_t)t * LAT + k];
      pk += aminoVec[k] * tp_w[(size_t)t * LAT + k];
    }
    float fc_ = mu_theta[(size_t)c_id[0] * KHEAD + t] * ck;
    float fp_ = mu_beta [(size_t)p_id[0] * KHEAD + t] * pk;
    fusion[t]         = fc_ > 0.f ? fc_ : 0.1f * fc_;
    fusion[KHEAD + t] = fp_ > 0.f ? fp_ : 0.1f * fp_;
  }
  {
    float s = 0.f;
    for (int k = 0; k < 1024; ++k) s += fps[k] * fp0[(size_t)k * LAT + t];
    fv1[t] = s > 0.f ? s : 0.1f * s;
  }
  __syncthreads();
  if (t < KHEAD) {
    float s = 0.f;
    for (int k = 0; k < LAT; ++k) s += fv1[k] * fp1[(size_t)k * KHEAD + t];
    fusion[2 * KHEAD + t] = s > 0.f ? s : 0.1f * s;
  }
  __syncthreads();
  if (t < 2) {
    float s = out_b[t];
    for (int i = 0; i < 3 * KHEAD; ++i) s += fusion[i] * out_w[(size_t)t * (3 * KHEAD) + i];
    out[t] = s;
  }
}

// ---------------------------------------------------------------------------
// Orchestration
// ---------------------------------------------------------------------------
enum {
  IN_FPS = 0, IN_ATOM_EMB, IN_AMINO_EMB,
  IN_G0_W1, IN_G0_B1, IN_G0_W2, IN_G0_B2,
  IN_G1_W1, IN_G1_B1, IN_G1_W2, IN_G1_B2,
  IN_G2_W1, IN_G2_B1, IN_G2_W2, IN_G2_B2,
  IN_P0_W, IN_P0_B, IN_P1_W, IN_P1_B, IN_P2_W, IN_P2_B, IN_P3_W, IN_P3_B,
  IN_FC_W, IN_FC_B,
  IN_C0_W, IN_C0_B, IN_C1_W, IN_C1_B, IN_C2_W, IN_C2_B,
  IN_TC_W, IN_TC_B, IN_TP_W, IN_TP_B,
  IN_FP0, IN_FP1, IN_MU_T, IN_MU_B,
  IN_OUT_W, IN_OUT_B,
  IN_ATOMS, IN_ESRC, IN_EDST, IN_AMINO, IN_CID, IN_PID
};

extern "C" void kernel_launch(void* const* d_in, const int* in_sizes, int n_in,
                              void* d_out, int out_size, void* d_ws, size_t ws_size,
                              hipStream_t stream) {
  (void)in_sizes; (void)n_in; (void)out_size; (void)ws_size;
  auto F  = [&](int i) { return (const float*)d_in[i]; };
  auto I  = [&](int i) { return (const int*)d_in[i]; };

  // ---- workspace carve (256B aligned) ----
  char* wp = (char*)d_ws;
  auto carve = [&](size_t bytes) -> void* {
    void* p = wp; wp += (bytes + 255) & ~(size_t)255; return p;
  };
  float*          hF      = (float*)         carve((size_t)NN * HID * 4);
  float*          zF      = (float*)         carve((size_t)NN * HID * 4);
  float*          mF      = (float*)         carve((size_t)NN * HID * 4);
  unsigned short* zB      = (unsigned short*)carve((size_t)NN * HID * 2);
  unsigned short* aB      = (unsigned short*)carve((size_t)LP * PROT * 2);
  unsigned short* wB      = (unsigned short*)carve((size_t)KSZ * 1024 * 512 * 2);
  float*          xF      = (float*)         carve((size_t)LP * LAT * 4);
  float*          yF      = (float*)         carve((size_t)LP * 2 * LAT * 4);
  unsigned short* xPadB   = (unsigned short*)carve((size_t)(LP + 16) * LAT * 2);
  float*          pooled  = (float*)         carve(4 * LAT * 4);
  float*          score   = (float*)         carve(LAT * 4);
  float*          aminoV  = (float*)         carve(LAT * 4);

  const int EW = 4096;  // grid-stride elementwise blocks

  // ======== compound branch: embedding + GIN ========
  gather_rows_f32_k<<<NN, 256, 0, stream>>>(F(IN_ATOM_EMB), I(IN_ATOMS), hF, COMP);
  colsum_k<<<COMP, 256, 0, stream>>>(hF, pooled + 0 * LAT, NN, COMP);

  const int w1i[3] = {IN_G0_W1, IN_G1_W1, IN_G2_W1};
  const int b1i[3] = {IN_G0_B1, IN_G1_B1, IN_G2_B1};
  const int w2i[3] = {IN_G0_W2, IN_G1_W2, IN_G2_W2};
  const int b2i[3] = {IN_G0_B2, IN_G1_B2, IN_G2_B2};

  for (int i = 0; i < 3; ++i) {
    int Din = (i == 0) ? COMP : HID;
    // z = h + segment_sum(h[src] at dst)
    copy_f32_k<<<EW, 256, 0, stream>>>(hF, zF, (size_t)NN * Din);
    edge_scatter_k<<<NE, 256, 0, stream>>>(hF, zF, I(IN_ESRC), I(IN_EDST), Din);
    f32_to_bf16_k<<<EW, 256, 0, stream>>>(zF, zB, (size_t)NN * Din);
    // m = relu(inorm(z @ W1^T + b1))
    f32_to_bf16_k<<<EW, 256, 0, stream>>>(F(w1i[i]), wB, (size_t)HID * Din);
    wmma_gemm_bf16<<<dim3(HID / 64, NN / 128), 256, 0, stream>>>(
        zB, Din, wB, Din, mF, HID, F(b1i[i]), Din, 0);
    inorm_relu_k<<<NN / 8, 256, 0, stream>>>(mF, nullptr, zB, HID, 0);
    // h = relu(inorm(relu(inorm(m @ W2^T + b2))))
    f32_to_bf16_k<<<EW, 256, 0, stream>>>(F(w2i[i]), wB, (size_t)HID * HID);
    wmma_gemm_bf16<<<dim3(HID / 64, NN / 128), 256, 0, stream>>>(
        zB, HID, wB, HID, zF, HID, F(b2i[i]), HID, 0);
    inorm_relu_k<<<NN / 8, 256, 0, stream>>>(zF, hF, nullptr, HID, 1);
    colsum_k<<<HID, 256, 0, stream>>>(hF, pooled + (size_t)(i + 1) * LAT, NN, HID);
  }
  // score = sum_i pooled_i @ p_i^T + p_i_b
  matvec_k<<<(LAT + 255) / 256, 256, 0, stream>>>(pooled + 0 * LAT, F(IN_P0_W), F(IN_P0_B), score, COMP, LAT, 1);
  matvec_k<<<(LAT + 255) / 256, 256, 0, stream>>>(pooled + 1 * LAT, F(IN_P1_W), F(IN_P1_B), score, HID,  LAT, 0);
  matvec_k<<<(LAT + 255) / 256, 256, 0, stream>>>(pooled + 2 * LAT, F(IN_P2_W), F(IN_P2_B), score, HID,  LAT, 0);
  matvec_k<<<(LAT + 255) / 256, 256, 0, stream>>>(pooled + 3 * LAT, F(IN_P3_W), F(IN_P3_B), score, HID,  LAT, 0);

  // ======== protein branch: embedding + fc + GLU conv stack ========
  gather_rows_bf16_k<<<LP, 256, 0, stream>>>(F(IN_AMINO_EMB), I(IN_AMINO), aB, PROT);
  f32_to_bf16_k<<<EW, 256, 0, stream>>>(F(IN_FC_W), wB, (size_t)LAT * PROT);
  wmma_gemm_bf16<<<dim3(LAT / 64, LP / 128), 256, 0, stream>>>(
      aB, PROT, wB, PROT, xF, LAT, F(IN_FC_B), PROT, 0);

  // padded bf16 activations: rows [0,5) and [LP+5, LP+16) zero, data at +5 rows
  zero_u16_k<<<64, 256, 0, stream>>>(xPadB, (size_t)CPAD * LAT);
  zero_u16_k<<<64, 256, 0, stream>>>(xPadB + (size_t)(LP + CPAD) * LAT, (size_t)11 * LAT);
  f32_to_bf16_k<<<EW, 256, 0, stream>>>(xF, xPadB + (size_t)CPAD * LAT, (size_t)LP * LAT);

  const int cwi[3] = {IN_C0_W, IN_C1_W, IN_C2_W};
  const int cbi[3] = {IN_C0_B, IN_C1_B, IN_C2_B};
  for (int i = 0; i < 3; ++i) {
    // repack conv weights [co][ci][k] f32 -> per-tap [k][co][ci] bf16
    pack_conv_k<<<EW, 256, 0, stream>>>(F(cwi[i]), wB, 2 * LAT, LAT, KSZ);
    // conv1d as 11 shifted GEMMs: y[l,co] (+)= xpad[l+k,:] . Wtap_k[co,:]
    for (int k = 0; k < KSZ; ++k) {
      wmma_gemm_bf16<<<dim3((2 * LAT) / 64, LP / 128), 256, 0, stream>>>(
          xPadB + (size_t)k * LAT, LAT,
          wB + (size_t)k * (2 * LAT) * LAT, LAT,
          yF, 2 * LAT,
          (k == 0) ? F(cbi[i]) : nullptr, LAT, (k > 0) ? 1 : 0);
    }
    glu_residual_k<<<LP, 256, 0, stream>>>(yF, xF, xPadB);
  }
  // amino_vec = sum_l inorm(leaky_relu(x[l,:], 0.1))
  zero_f32_k<<<1, 256, 0, stream>>>(aminoV, LAT);
  leaky_inorm_colsum_k<<<LP / 8, 256, 0, stream>>>(xF, aminoV, LAT);

  // ======== fusion head ========
  head_k<<<1, 512, 0, stream>>>(score, aminoV, F(IN_FPS),
                                F(IN_TC_W), F(IN_TC_B), F(IN_TP_W), F(IN_TP_B),
                                F(IN_MU_T), F(IN_MU_B), F(IN_FP0), F(IN_FP1),
                                F(IN_OUT_W), F(IN_OUT_B),
                                I(IN_CID), I(IN_PID), (float*)d_out);
}
